// GroupAttention_70308614636049
// MI455X (gfx1250) — compile-verified, matrix operand order checked by reading, not run on hardware
//
#include <hip/hip_runtime.h>
#include <hip/hip_bf16.h>
#include <math.h>
#include <stdint.h>

typedef __attribute__((ext_vector_type(16))) __bf16 v16bf;
typedef __attribute__((ext_vector_type(8)))  float  v8f;
typedef __attribute__((ext_vector_type(4)))  unsigned int v4u;
typedef __attribute__((ext_vector_type(8)))  int v8i;
typedef __attribute__((ext_vector_type(4)))  int v4i;

#define B_ 8
#define S_ 1024
#define H_ 1024
#define LN_EPS 1e-5f

// ---------------- LayerNorm -> bf16 ----------------
__global__ __launch_bounds__(256) void ln_kernel(const float* __restrict__ x,
                                                 const float* __restrict__ gamma,
                                                 const float* __restrict__ beta,
                                                 __bf16* __restrict__ cn) {
  int row = blockIdx.x;              // 0..B*S-1
  int tid = threadIdx.x;             // 256 threads, 4 elems each
  const float* xr = x + (size_t)row * H_;
  float v[4];
  float s = 0.f, s2 = 0.f;
  for (int i = 0; i < 4; ++i) {
    float t = xr[tid + i * 256];
    v[i] = t; s += t; s2 += t * t;
  }
  __shared__ float rs[256], rs2[256];
  rs[tid] = s; rs2[tid] = s2;
  __syncthreads();
  for (int st = 128; st > 0; st >>= 1) {
    if (tid < st) { rs[tid] += rs[tid + st]; rs2[tid] += rs2[tid + st]; }
    __syncthreads();
  }
  float mean = rs[0] * (1.0f / H_);
  float var  = rs2[0] * (1.0f / H_) - mean * mean;
  float rstd = rsqrtf(var + LN_EPS);
  __bf16* cr = cn + (size_t)row * H_;
  for (int i = 0; i < 4; ++i) {
    int c = tid + i * 256;
    float y = (v[i] - mean) * rstd * gamma[c] + beta[c];
    cr[c] = (__bf16)y;
  }
}

// ---------------- f32 -> bf16 weight convert ----------------
__global__ void cvt_bf16_kernel(const float* __restrict__ in, __bf16* __restrict__ out, int n) {
  int i = blockIdx.x * blockDim.x + threadIdx.x;
  if (i < n) out[i] = (__bf16)in[i];
}

// ---------------- bf16 WMMA GEMM: C[M,N] = A[M,K] * B[K,N] + bias ----------------
#define BM 128
#define BN 128
#define BK 32
#define LDP 40   // padded LDS row: 32 bf16 data + 8 bf16 pad = 80 B (16B aligned, bank friendly)

__global__ __launch_bounds__(256) void gemm_bf16_kernel(const __bf16* __restrict__ A,
                                                        const __bf16* __restrict__ Bm,
                                                        const float* __restrict__ bias,
                                                        float* __restrict__ C,
                                                        int M, int N, int K) {
  __shared__ __bf16 As[BM][LDP];    // A tile, row-major, padded (filled by TDM)
  __shared__ __bf16 Bst[BN][LDP];   // B tile TRANSPOSED: [n][k], padded
  int tid  = threadIdx.x;
  int lane = tid & 31;
  int wid  = tid >> 5;              // 8 waves
  int wm   = wid & 3;               // 0..3  -> 32-row strip
  int wn   = wid >> 2;              // 0..1  -> 64-col strip
  int m0 = blockIdx.y * BM;
  int n0 = blockIdx.x * BN;
  int hi  = lane >> 4;              // half-wave
  int l15 = lane & 15;

  v8f acc[2][4] = {};

  for (int k0 = 0; k0 < K; k0 += BK) {
    // ---- TDM: DMA the A tile (BM x BK bf16) into padded LDS, wave 0 issues ----
    if (wid == 0) {
      unsigned lds_off = (unsigned)(uintptr_t)&As[0][0];
      unsigned long long ga =
          (unsigned long long)(uintptr_t)A + ((unsigned long long)m0 * K + k0) * 2ull;
      v4u g0;
      g0[0] = 1u;                                   // count=1 valid descriptor
      g0[1] = lds_off;                              // lds_addr
      g0[2] = (unsigned)ga;                         // global_addr[31:0]
      g0[3] = (unsigned)((ga >> 32) & 0x01FFFFFFu)  // global_addr[56:32]
              | 0x80000000u;                        // type=2 ("image")
      v8i g1;
      g1[0] = (1 << 16)      // data_size = 2 bytes
            | (1 << 20)      // pad_enable
            | (3 << 22)      // pad_interval: 16 DWORDs (one 64B tile row)
            | (3 << 25);     // pad_amount:   4 DWORDs (16B -> row stride 80B)
      g1[1] = (int)(((unsigned)K & 0xFFFFu) << 16);          // tensor_dim0 lo16
      g1[2] = (int)((((unsigned)K >> 16) & 0xFFFFu)          // tensor_dim0 hi16
            | (((unsigned)M & 0xFFFFu) << 16));              // tensor_dim1 lo16
      g1[3] = (int)((((unsigned)M >> 16) & 0xFFFFu)          // tensor_dim1 hi16
            | (BK << 16));                                   // tile_dim0 = 32
      g1[4] = BM;                                            // tile_dim1 = 128, tile_dim2 = 0
      g1[5] = K;                                             // tensor_dim0_stride lo32
      g1[6] = 0;
      g1[7] = 0;
      v4i gz4 = {0, 0, 0, 0};
      v8i gz8 = {0, 0, 0, 0, 0, 0, 0, 0};
      __builtin_amdgcn_tensor_load_to_lds(g0, g1, gz4, gz4, gz8, 0);
    }

    // ---- stage B tile transposed: Bst[n][k] (all 256 threads) ----
    // 512 chunks of 8 contiguous bf16 along N; coalesced 16B global loads
    for (int p = tid; p < (BK * BN) / 8; p += 256) {
      int r  = p >> 4;            // k row 0..31
      int c8 = (p & 15) << 3;     // col start 0..120
      union { v4u q; __bf16 h[8]; } u;
      u.q = *(const v4u*)&Bm[(size_t)(k0 + r) * N + n0 + c8];
      if (k0 + BK < K)  // prefetch next K-tile of B
        __builtin_prefetch(&Bm[(size_t)(k0 + BK + r) * N + n0 + c8], 0, 0);
      for (int j = 0; j < 8; ++j) Bst[c8 + j][r] = u.h[j];
    }

    if (wid == 0) __builtin_amdgcn_s_wait_tensorcnt((short)0);
    __syncthreads();

    // ---- fragments: two contiguous 16B chunks per lane (K 0..7 / 16..23 per half) ----
    union { v16bf v; v4u q[2]; } af[2], bfr[4];
    for (int mt = 0; mt < 2; ++mt) {
      const __bf16* ap = &As[wm * 32 + mt * 16 + l15][0];
      af[mt].q[0] = *(const v4u*)(ap + 8 * hi);
      af[mt].q[1] = *(const v4u*)(ap + 16 + 8 * hi);
    }
    for (int nt = 0; nt < 4; ++nt) {
      const __bf16* bp = &Bst[wn * 64 + nt * 16 + l15][0];
      bfr[nt].q[0] = *(const v4u*)(bp + 8 * hi);
      bfr[nt].q[1] = *(const v4u*)(bp + 16 + 8 * hi);
    }
    for (int mt = 0; mt < 2; ++mt)
      for (int nt = 0; nt < 4; ++nt)
        acc[mt][nt] = __builtin_amdgcn_wmma_f32_16x16x32_bf16(
            false, af[mt].v, false, bfr[nt].v, (short)0, acc[mt][nt], false, false);
    __syncthreads();
  }

  // Epilogue: D layout — VGPR r holds (M = r + 8*hi, N = lane&15)
  for (int mt = 0; mt < 2; ++mt)
    for (int nt = 0; nt < 4; ++nt) {
      int col = n0 + wn * 64 + nt * 16 + l15;
      float bv = bias[col];
      for (int r = 0; r < 8; ++r) {
        int row = m0 + wm * 32 + mt * 16 + r + (hi << 3);
        C[(size_t)row * N + col] = acc[mt][nt][r] + bv;
      }
    }
}

// ---------------- neighbor scores + masked 2-way softmax ----------------
__global__ __launch_bounds__(256) void score_kernel(const float* __restrict__ Q,
                                                    const float* __restrict__ Kv,
                                                    const int* __restrict__ mask,
                                                    float* __restrict__ sup,
                                                    float* __restrict__ sdn,
                                                    float* __restrict__ uu) {
  int row = blockIdx.x;                 // b*S + s
  int b = row >> 10, s = row & (S_ - 1);
  int tid = threadIdx.x;
  const float* qr = Q + (size_t)row * H_;
  float du = 0.f, dd = 0.f;
  if (s + 1 < S_) {
    const float* kr = Kv + (size_t)(row + 1) * H_;
    for (int i = tid; i < H_; i += 256) du += qr[i] * kr[i];
  }
  if (s > 0) {
    const float* kr = Kv + (size_t)(row - 1) * H_;
    for (int i = tid; i < H_; i += 256) dd += qr[i] * kr[i];
  }
  __shared__ float r1[256], r2[256];
  r1[tid] = du; r2[tid] = dd;
  __syncthreads();
  for (int st = 128; st > 0; st >>= 1) {
    if (tid < st) { r1[tid] += r1[tid + st]; r2[tid] += r2[tid + st]; }
    __syncthreads();
  }
  if (tid == 0) {
    const float scale = 0.03125f;      // 1/sqrt(1024)
    bool p  = mask[b * S_ + s] != 0;
    bool up = (s + 1 < S_) && p && (mask[b * S_ + s + 1] != 0);
    bool dn = (s > 0)      && p && (mask[b * S_ + s - 1] != 0);
    float suv, sdv, uv;
    if (up || dn) {
      float a = up ? r1[0] * scale : -1e30f;
      float c = dn ? r2[0] * scale : -1e30f;
      float mx = fmaxf(a, c);
      float eu = up ? expf(a - mx) : 0.f;
      float ed = dn ? expf(c - mx) : 0.f;
      float z = eu + ed;
      suv = eu / z; sdv = ed / z; uv = 0.f;
    } else {
      suv = sdv = uv = 1.0f / S_;      // all scores -1e9 -> uniform softmax
    }
    sup[row] = suv; sdn[row] = sdv; uu[row] = uv;
  }
}

__device__ __forceinline__ float m_val(int s, int c, int row, int b,
                                       const float* sup, const float* sdn,
                                       const float* uu, float ur) {
  if (c == s + 1)      return sqrtf(sup[row] * sdn[row + 1] + 1e-9f);
  else if (c == s - 1) return sqrtf(sdn[row] * sup[row - 1] + 1e-9f);
  else                 return sqrtf(ur * uu[b * S_ + c] + 1e-9f);
}

// ---------------- per-row normalizer: sum_{c!=s} (prior + (1-prior)*m) ----------------
__global__ __launch_bounds__(256) void rowsum_kernel(const float* __restrict__ prior,
                                                     const float* __restrict__ sup,
                                                     const float* __restrict__ sdn,
                                                     const float* __restrict__ uu,
                                                     float* __restrict__ dinv) {
  int row = blockIdx.x;
  int b = row >> 10, s = row & (S_ - 1);
  int tid = threadIdx.x;
  const float4* pr = (const float4*)(prior + (size_t)b * S_ * S_ + (size_t)s * S_);
  float ur = uu[row];
  int c0 = tid << 2;                       // 256 threads x 4 = 1024 cols, one shot
  float4 p4 = pr[tid];
  float t = 0.f;
  {
    float pv[4] = {p4.x, p4.y, p4.z, p4.w};
    for (int j = 0; j < 4; ++j) {
      int c = c0 + j;
      if (c == s) continue;
      float m = m_val(s, c, row, b, sup, sdn, uu, ur);
      t += pv[j] + (1.f - pv[j]) * m;
    }
  }
  __shared__ float r1[256];
  r1[tid] = t;
  __syncthreads();
  for (int st = 128; st > 0; st >>= 1) {
    if (tid < st) r1[tid] += r1[tid + st];
    __syncthreads();
  }
  // rowsum(g_pre) + 1e-9 = (S-1) + T + (2 + 1e-9) + 1e-9
  if (tid == 0) dinv[row] = 1.f / ((float)(S_ + 1) + r1[0] + 2e-9f);
}

// ---------------- fused outputs: g and neibor ----------------
__global__ __launch_bounds__(256) void out_kernel(const float* __restrict__ prior,
                                                  const int* __restrict__ mask,
                                                  const float* __restrict__ sup,
                                                  const float* __restrict__ sdn,
                                                  const float* __restrict__ uu,
                                                  const float* __restrict__ dinv,
                                                  float* __restrict__ g,
                                                  float* __restrict__ nb) {
  int row = blockIdx.x;
  int b = row >> 10, s = row & (S_ - 1);
  int tid = threadIdx.x;
  size_t base = (size_t)b * S_ * S_ + (size_t)s * S_;
  float ur = uu[row];
  float di = dinv[row];
  bool ps = mask[b * S_ + s] != 0;
  int c0 = tid << 2;
  float4 p4 = *(const float4*)(prior + base + c0);
  float pv[4] = {p4.x, p4.y, p4.z, p4.w};
  float nv[4], gv[4];
  for (int j = 0; j < 4; ++j) {
    int c = c0 + j;
    float m = m_val(s, c, row, b, sup, sdn, uu, ur);   // diag: sqrt(ur^2+1e-9)
    float N = pv[j] + (1.f - pv[j]) * m;
    float f = (ps && (mask[b * S_ + c] != 0)) ? 1.f : 0.f;
    nv[j] = N * f;
    float gp = (c == s) ? (2.f + 1e-9f) : (1.f + N);
    gv[j] = gp * di * f;
  }
  *(float4*)(nb + base + c0) = make_float4(nv[0], nv[1], nv[2], nv[3]);
  *(float4*)(g  + base + c0) = make_float4(gv[0], gv[1], gv[2], gv[3]);
}

extern "C" void kernel_launch(void* const* d_in, const int* in_sizes, int n_in,
                              void* d_out, int out_size, void* d_ws, size_t ws_size,
                              hipStream_t stream) {
  const float* context = (const float*)d_in[0];
  const int*   mask    = (const int*)d_in[1];
  const float* prior   = (const float*)d_in[2];
  const float* gamma   = (const float*)d_in[3];
  const float* beta    = (const float*)d_in[4];
  const float* Wk      = (const float*)d_in[5];
  const float* bk      = (const float*)d_in[6];
  const float* Wq      = (const float*)d_in[7];
  const float* bq      = (const float*)d_in[8];

  float* g  = (float*)d_out;
  float* nb = g + (size_t)B_ * S_ * S_;

  char* ws = (char*)d_ws;
  __bf16* CN  = (__bf16*)ws;  ws += (size_t)B_ * S_ * H_ * 2;
  __bf16* Wqb = (__bf16*)ws;  ws += (size_t)H_ * H_ * 2;
  __bf16* Wkb = (__bf16*)ws;  ws += (size_t)H_ * H_ * 2;
  float*  Q   = (float*)ws;   ws += (size_t)B_ * S_ * H_ * 4;
  float*  Kt  = (float*)ws;   ws += (size_t)B_ * S_ * H_ * 4;
  float*  sup = (float*)ws;   ws += (size_t)B_ * S_ * 4;
  float*  sdn = (float*)ws;   ws += (size_t)B_ * S_ * 4;
  float*  uu  = (float*)ws;   ws += (size_t)B_ * S_ * 4;
  float*  dinv= (float*)ws;   ws += (size_t)B_ * S_ * 4;

  ln_kernel<<<B_ * S_, 256, 0, stream>>>(context, gamma, beta, CN);
  cvt_bf16_kernel<<<(H_ * H_ + 255) / 256, 256, 0, stream>>>(Wq, Wqb, H_ * H_);
  cvt_bf16_kernel<<<(H_ * H_ + 255) / 256, 256, 0, stream>>>(Wk, Wkb, H_ * H_);

  dim3 ggrid(H_ / BN, (B_ * S_) / BM);
  gemm_bf16_kernel<<<ggrid, 256, 0, stream>>>(CN, Wqb, bq, Q, B_ * S_, H_, H_);
  gemm_bf16_kernel<<<ggrid, 256, 0, stream>>>(CN, Wkb, bk, Kt, B_ * S_, H_, H_);

  score_kernel<<<B_ * S_, 256, 0, stream>>>(Q, Kt, mask, sup, sdn, uu);
  rowsum_kernel<<<B_ * S_, 256, 0, stream>>>(prior, sup, sdn, uu, dinv);
  out_kernel<<<B_ * S_, 256, 0, stream>>>(prior, mask, sup, sdn, uu, dinv, g, nb);
}
